// AttentionFlow_Layer_17360257810679
// MI455X (gfx1250) — compile-verified, matrix operand order checked by reading, not run on hardware
//
#include <hip/hip_runtime.h>
#include <hip/hip_bf16.h>
#include <math.h>

typedef __bf16 bf16_t;
typedef __attribute__((ext_vector_type(16))) __bf16 v16bf;
typedef __attribute__((ext_vector_type(8)))  __bf16 v8bf;
typedef __attribute__((ext_vector_type(8)))  float  v8f;

#define WMMA_BF16(a_, b_, c_) \
  __builtin_amdgcn_wmma_f32_16x16x32_bf16(false, (a_), false, (b_), (short)0, (c_), false, false)

constexpr int BATCH = 32, LC = 1024, LQ = 128, D = 256;
constexpr float NEG_INF = -1e10f;

// workspace layout (bytes); total ~53 MB
constexpr size_t OFF_CTXBF = 0;                                        // bf16 ctx   [B][LC][D]
constexpr size_t OFF_QW    = OFF_CTXBF + (size_t)BATCH * LC * D * 2;   // bf16 q*ww  [B][LQ][D]
constexpr size_t OFF_QT    = OFF_QW    + (size_t)BATCH * LQ * D * 2;   // bf16 q^T   [B][D][LQ]
constexpr size_t OFF_WT    = OFF_QT    + (size_t)BATCH * LQ * D * 2;   // bf16 W^T   [D][4D]
constexpr size_t OFF_WC    = OFF_WT    + (size_t)(4 * D) * D * 2;      // f32 ctx.wc [B][LC]
constexpr size_t OFF_WQ    = OFF_WC    + (size_t)BATCH * LC * 4;       // f32 q.wq   [B][LQ]
constexpr size_t OFF_M     = OFF_WQ    + (size_t)BATCH * LQ * 4;       // f32 rowmax [B][LC]
constexpr size_t OFF_Q2C   = OFF_M     + (size_t)BATCH * LC * 4;       // f32 q2c    [B][D]
constexpr size_t OFF_C2Q   = OFF_Q2C   + (size_t)BATCH * D * 4;        // f32 c2q    [B][LC][D]

__device__ inline v8f zero8() {
  v8f z;
#pragma unroll
  for (int t = 0; t < 8; ++t) z[t] = 0.f;
  return z;
}

// A/B fragment from K-major bf16 storage: lo = 8 halves at k, hi = 8 halves at k+16
__device__ inline v16bf make_frag(const bf16_t* lo, const bf16_t* hi) {
  v8bf l = *(const v8bf*)lo;
  v8bf h = *(const v8bf*)hi;
  v16bf r;
#pragma unroll
  for (int t = 0; t < 8; ++t) { r[t] = l[t]; r[t + 8] = h[t]; }
  return r;
}

__device__ inline v16bf cvt_frag(const float* lo, const float* hi) {
  v8f l = *(const v8f*)lo;
  v8f h = *(const v8f*)hi;
  v16bf r;
#pragma unroll
  for (int t = 0; t < 8; ++t) { r[t] = (bf16_t)l[t]; r[t + 8] = (bf16_t)h[t]; }
  return r;
}

__device__ inline v16bf mul_frag(const float* x, const float* y) {
  v8f xl = *(const v8f*)x, xh = *(const v8f*)(x + 16);
  v8f yl = *(const v8f*)y, yh = *(const v8f*)(y + 16);
  v16bf r;
#pragma unroll
  for (int t = 0; t < 8; ++t) {
    r[t]     = (bf16_t)(xl[t] * yl[t]);
    r[t + 8] = (bf16_t)(xh[t] * yh[t]);
  }
  return r;
}

// ---------------- prep kernels ----------------

__global__ void __launch_bounds__(256) k_prep_w(const float* __restrict__ W,
                                                bf16_t* __restrict__ Wt) {
  int idx = blockIdx.x * 256 + threadIdx.x;   // 0 .. 4*D*D-1
  int k = idx >> 8;        // row in merge_W (0..1023)
  int n = idx & 255;       // col (0..255)
  Wt[(size_t)n * (4 * D) + k] = (bf16_t)W[idx];
}

__global__ void __launch_bounds__(256) k_prep_q(const float* __restrict__ q,
                                                const float* __restrict__ w_sim,
                                                bf16_t* __restrict__ qw,
                                                bf16_t* __restrict__ qT,
                                                float* __restrict__ wqdot) {
  const int b = blockIdx.y, j = blockIdx.x, t = threadIdx.x;  // t = d
  const float v = q[((size_t)b * LQ + j) * D + t];
  qw[((size_t)b * LQ + j) * D + t] = (bf16_t)(v * w_sim[2 * D + t]);
  qT[((size_t)b * D + t) * LQ + j] = (bf16_t)v;
  float p = v * w_sim[D + t];
#pragma unroll
  for (int off = 16; off; off >>= 1) p += __shfl_xor(p, off, 32);
  __shared__ float red[8];
  if ((t & 31) == 0) red[t >> 5] = p;
  __syncthreads();
  if (t == 0) {
    float s = 0.f;
#pragma unroll
    for (int i = 0; i < 8; ++i) s += red[i];
    wqdot[b * LQ + j] = s;
  }
}

__global__ void __launch_bounds__(256) k_prep_ctx(const float* __restrict__ ctx,
                                                  const float* __restrict__ w_sim,
                                                  bf16_t* __restrict__ ctxb,
                                                  float* __restrict__ wcdot) {
  const int b = blockIdx.y, i = blockIdx.x, t = threadIdx.x;
  const float v = ctx[((size_t)b * LC + i) * D + t];
  ctxb[((size_t)b * LC + i) * D + t] = (bf16_t)v;
  float p = v * w_sim[t];
#pragma unroll
  for (int off = 16; off; off >>= 1) p += __shfl_xor(p, off, 32);
  __shared__ float red[8];
  if ((t & 31) == 0) red[t >> 5] = p;
  __syncthreads();
  if (t == 0) {
    float s = 0.f;
#pragma unroll
    for (int i2 = 0; i2 < 8; ++i2) s += red[i2];
    wcdot[b * LC + i] = s;
  }
}

// ---------------- scores + softmax + c2q ----------------
// block = 256 threads = 8 waves; tile = 128 ctx rows x all 128 queries
__global__ void __launch_bounds__(256) k_scores(const bf16_t* __restrict__ ctxb,
                                                const bf16_t* __restrict__ qw,
                                                const bf16_t* __restrict__ qT,
                                                const float* __restrict__ wcdot,
                                                const float* __restrict__ wqdot,
                                                const unsigned char* __restrict__ cmask,
                                                const unsigned char* __restrict__ qmask,
                                                float* __restrict__ m_out,
                                                float* __restrict__ c2q) {
  const int b  = blockIdx.y;
  const int i0 = blockIdx.x * 128;
  const int tid = threadIdx.x, wv = tid >> 5, ln = tid & 31;
  const int lnm = ln & 15, sel = ln >> 4;
  constexpr int PS = 144;  // LDS row stride (halves): 288 B, 16 B-aligned rows

  __shared__ float wc_s[128], wq_s[128];
  __shared__ unsigned char cm_s[128], qm_s[128];
  __shared__ __align__(16) bf16_t P[128 * PS];

  if (tid < 128) {
    wc_s[tid] = wcdot[b * LC + i0 + tid];
    cm_s[tid] = cmask[(size_t)b * LC + i0 + tid];
    wq_s[tid] = wqdot[b * LQ + tid];
    qm_s[tid] = qmask[(size_t)b * LQ + tid];
  }
  __syncthreads();

  // ---- S = ctx_bf16 @ (q*ww)^T, K = 256 ----
  v8f acc[8];
#pragma unroll
  for (int nt = 0; nt < 8; ++nt) acc[nt] = zero8();
  const bf16_t* abase = ctxb + ((size_t)b * LC + i0 + wv * 16 + lnm) * D;
  for (int k0 = 0; k0 < D; k0 += 32) {
    v16bf a = make_frag(abase + k0 + sel * 8, abase + k0 + sel * 8 + 16);
#pragma unroll
    for (int nt = 0; nt < 8; ++nt) {
      const int j = nt * 16 + lnm;
      const bf16_t* bb = qw + ((size_t)b * LQ + j) * D + k0 + sel * 8;
      v16bf bfrag = make_frag(bb, bb + 16);
      acc[nt] = WMMA_BF16(a, bfrag, acc[nt]);
    }
  }

  // ---- bias + mask ----
#pragma unroll
  for (int nt = 0; nt < 8; ++nt) {
    const int n = nt * 16 + lnm;
    const bool qm = qm_s[n] != 0;
    const float wqv = wq_s[n];
#pragma unroll
    for (int r = 0; r < 8; ++r) {
      const int lr = wv * 16 + r + 8 * sel;
      float s = acc[nt][r] + wc_s[lr] + wqv;
      acc[nt][r] = (qm || cm_s[lr]) ? NEG_INF : s;
    }
  }

  // ---- row max (over all 128 j) + softmax ----
  float rmax[8], rinv[8];
#pragma unroll
  for (int r = 0; r < 8; ++r) {
    float mr = acc[0][r];
#pragma unroll
    for (int nt = 1; nt < 8; ++nt) mr = fmaxf(mr, acc[nt][r]);
    mr = fmaxf(mr, __shfl_xor(mr, 1, 32));
    mr = fmaxf(mr, __shfl_xor(mr, 2, 32));
    mr = fmaxf(mr, __shfl_xor(mr, 4, 32));
    mr = fmaxf(mr, __shfl_xor(mr, 8, 32));
    rmax[r] = mr;
    float s = 0.f;
#pragma unroll
    for (int nt = 0; nt < 8; ++nt) {
      float e = __expf(acc[nt][r] - mr);
      acc[nt][r] = e;
      s += e;
    }
    s += __shfl_xor(s, 1, 32);
    s += __shfl_xor(s, 2, 32);
    s += __shfl_xor(s, 4, 32);
    s += __shfl_xor(s, 8, 32);
    rinv[r] = 1.f / s;
  }
  if (lnm == 0) {
#pragma unroll
    for (int r = 0; r < 8; ++r)
      m_out[(size_t)b * LC + i0 + wv * 16 + r + 8 * sel] = rmax[r];
  }

  // ---- P -> LDS (C layout -> memory layout for A-frag reload) ----
#pragma unroll
  for (int nt = 0; nt < 8; ++nt) {
    const int n = nt * 16 + lnm;
#pragma unroll
    for (int r = 0; r < 8; ++r) {
      const int lr = wv * 16 + r + 8 * sel;
      P[lr * PS + n] = (bf16_t)(acc[nt][r] * rinv[r]);
    }
  }
  __syncthreads();

  // ---- c2q = P (rows) @ q, K = 128, N = 256 in two halves ----
  const bf16_t* prow = P + (wv * 16 + lnm) * PS;
#pragma unroll
  for (int h = 0; h < 2; ++h) {
    v8f cacc[8];
#pragma unroll
    for (int nt = 0; nt < 8; ++nt) cacc[nt] = zero8();
    for (int k0 = 0; k0 < LQ; k0 += 32) {
      v16bf a = make_frag(prow + k0 + sel * 8, prow + k0 + sel * 8 + 16);
#pragma unroll
      for (int nt = 0; nt < 8; ++nt) {
        const int d0 = h * 128 + nt * 16 + lnm;
        const bf16_t* bb = qT + ((size_t)b * D + d0) * LQ + k0 + sel * 8;
        v16bf bfrag = make_frag(bb, bb + 16);
        cacc[nt] = WMMA_BF16(a, bfrag, cacc[nt]);
      }
    }
#pragma unroll
    for (int nt = 0; nt < 8; ++nt) {
      const int d0 = h * 128 + nt * 16 + lnm;
#pragma unroll
      for (int r = 0; r < 8; ++r) {
        const int lr = wv * 16 + r + 8 * sel;
        c2q[((size_t)b * LC + i0 + lr) * D + d0] = cacc[nt][r];
      }
    }
  }
}

// ---------------- beta softmax + q2c ----------------
__global__ void __launch_bounds__(256) k_beta(const float* __restrict__ m,
                                              const float* __restrict__ ctx,
                                              float* __restrict__ q2c) {
  const int b = blockIdx.x, t = threadIdx.x;
  __shared__ float ex[LC];
  __shared__ float red[8];
  __shared__ float bmax_s, bsum_s;

  float lm = -3.4e38f;
  for (int i = t; i < LC; i += 256) lm = fmaxf(lm, m[(size_t)b * LC + i]);
#pragma unroll
  for (int off = 16; off; off >>= 1) lm = fmaxf(lm, __shfl_xor(lm, off, 32));
  if ((t & 31) == 0) red[t >> 5] = lm;
  __syncthreads();
  if (t == 0) {
    float mx = red[0];
#pragma unroll
    for (int i = 1; i < 8; ++i) mx = fmaxf(mx, red[i]);
    bmax_s = mx;
  }
  __syncthreads();

  float ls = 0.f;
  for (int i = t; i < LC; i += 256) {
    float e = __expf(m[(size_t)b * LC + i] - bmax_s);
    ex[i] = e;
    ls += e;
  }
#pragma unroll
  for (int off = 16; off; off >>= 1) ls += __shfl_xor(ls, off, 32);
  if ((t & 31) == 0) red[t >> 5] = ls;
  __syncthreads();
  if (t == 0) {
    float s = 0.f;
#pragma unroll
    for (int i = 0; i < 8; ++i) s += red[i];
    bsum_s = s;
  }
  __syncthreads();

  const float inv = 1.f / bsum_s;
  float acc = 0.f;
  for (int i = 0; i < LC; ++i)
    acc += ex[i] * ctx[((size_t)b * LC + i) * D + t];
  q2c[b * D + t] = acc * inv;
}

// ---------------- fused final GEMM: relu(concat @ W + b) ----------------
// block: 64 rows x 128 cols; 8 waves = 4 row-strips x 2 col-strips
__global__ void __launch_bounds__(256) k_out(const bf16_t* __restrict__ ctxb,
                                             const float* __restrict__ ctxf,
                                             const float* __restrict__ c2q,
                                             const float* __restrict__ q2c,
                                             const bf16_t* __restrict__ Wt,
                                             const float* __restrict__ bias,
                                             const unsigned char* __restrict__ cmask,
                                             float* __restrict__ out) {
  const int b   = blockIdx.z;
  const int i0  = blockIdx.y * 64;
  const int nch = blockIdx.x;   // 0/1 -> cols [0,128) / [128,256)
  const int tid = threadIdx.x, wv = tid >> 5, ln = tid & 31;
  const int lnm = ln & 15, sel = ln >> 4;
  const int m0 = (wv & 3) * 16;
  const int n0 = nch * 128 + (wv >> 2) * 64;

  __shared__ float q2c_s[D];
  __shared__ unsigned char cm_s[64];
  if (tid < D) q2c_s[tid] = q2c[b * D + tid];
  if (tid < 64) cm_s[tid] = cmask[(size_t)b * LC + i0 + tid];
  __syncthreads();

  const int lr = m0 + lnm;
  const bool masked = cm_s[lr] != 0;
  const bf16_t* actx = ctxb + ((size_t)b * LC + i0 + lr) * D;
  const float* fctx  = ctxf + ((size_t)b * LC + i0 + lr) * D;
  const float* fc2q  = c2q  + ((size_t)b * LC + i0 + lr) * D;

  v8f acc[4];
#pragma unroll
  for (int nt = 0; nt < 4; ++nt) acc[nt] = zero8();

  for (int kk = 0; kk < 4 * D; kk += 32) {
    const int seg = kk >> 8;             // 0:ctx 1:c2q 2:ctx*c2q 3:ctx*q2c
    const int kl = (kk & 255) + sel * 8; // segment-local K base for this lane
    v16bf a;
    if (seg == 0) {
      a = make_frag(actx + kl, actx + kl + 16);
    } else if (seg == 1) {
      a = cvt_frag(fc2q + kl, fc2q + kl + 16);
    } else if (seg == 2) {
      a = mul_frag(fctx + kl, fc2q + kl);
    } else {
      v8f l = *(const v8f*)(fctx + kl);
      v8f h = *(const v8f*)(fctx + kl + 16);
#pragma unroll
      for (int t = 0; t < 8; ++t) {
        a[t]     = (bf16_t)(l[t] * q2c_s[kl + t]);
        a[t + 8] = (bf16_t)(h[t] * q2c_s[kl + 16 + t]);
      }
    }
    if (masked) {  // zero features for masked ctx rows -> out = relu(bias)
#pragma unroll
      for (int t = 0; t < 16; ++t) a[t] = (bf16_t)0.f;
    }
#pragma unroll
    for (int nt = 0; nt < 4; ++nt) {
      const int n = n0 + nt * 16 + lnm;
      const bf16_t* bb = Wt + (size_t)n * (4 * D) + kk + sel * 8;
      v16bf bfrag = make_frag(bb, bb + 16);
      acc[nt] = WMMA_BF16(a, bfrag, acc[nt]);
    }
  }

#pragma unroll
  for (int nt = 0; nt < 4; ++nt) {
    const int n = n0 + nt * 16 + lnm;
    const float bv = bias[n];
#pragma unroll
    for (int r = 0; r < 8; ++r) {
      const int orow = i0 + m0 + r + 8 * sel;
      float v = acc[nt][r] + bv;
      out[((size_t)b * LC + orow) * D + n] = v > 0.f ? v : 0.f;
    }
  }
}

extern "C" void kernel_launch(void* const* d_in, const int* in_sizes, int n_in,
                              void* d_out, int out_size, void* d_ws, size_t ws_size,
                              hipStream_t stream) {
  (void)in_sizes; (void)n_in; (void)out_size; (void)ws_size;
  const float* ctx            = (const float*)d_in[0];
  const unsigned char* cmask  = (const unsigned char*)d_in[1];
  const float* qry            = (const float*)d_in[2];
  const unsigned char* qmask  = (const unsigned char*)d_in[3];
  const float* w_sim          = (const float*)d_in[4];
  const float* mW             = (const float*)d_in[5];
  const float* mb             = (const float*)d_in[6];
  float* out = (float*)d_out;
  char* ws = (char*)d_ws;

  bf16_t* ctxb  = (bf16_t*)(ws + OFF_CTXBF);
  bf16_t* qw    = (bf16_t*)(ws + OFF_QW);
  bf16_t* qT    = (bf16_t*)(ws + OFF_QT);
  bf16_t* Wt    = (bf16_t*)(ws + OFF_WT);
  float*  wcdot = (float*)(ws + OFF_WC);
  float*  wqdot = (float*)(ws + OFF_WQ);
  float*  mrow  = (float*)(ws + OFF_M);
  float*  q2c   = (float*)(ws + OFF_Q2C);
  float*  c2q   = (float*)(ws + OFF_C2Q);

  k_prep_w  <<<dim3((4 * D * D) / 256), 256, 0, stream>>>(mW, Wt);
  k_prep_q  <<<dim3(LQ, BATCH),         256, 0, stream>>>(qry, w_sim, qw, qT, wqdot);
  k_prep_ctx<<<dim3(LC, BATCH),         256, 0, stream>>>(ctx, w_sim, ctxb, wcdot);
  k_scores  <<<dim3(LC / 128, BATCH),   256, 0, stream>>>(ctxb, qw, qT, wcdot, wqdot,
                                                          cmask, qmask, mrow, c2q);
  k_beta    <<<dim3(BATCH),             256, 0, stream>>>(mrow, ctx, q2c);
  k_out     <<<dim3(2, LC / 64, BATCH), 256, 0, stream>>>(ctxb, ctx, c2q, q2c, Wt, mb,
                                                          cmask, out);
}
